// MultiHeadAttention_47639777247377
// MI455X (gfx1250) — matrix operand order repaired and checked by reading, not compiled, not run on hardware
//
#include <hip/hip_runtime.h>
#include <hip/hip_bf16.h>

typedef __attribute__((ext_vector_type(16))) _Float16 v16h;
typedef __attribute__((ext_vector_type(8)))  _Float16 h8;
typedef __attribute__((ext_vector_type(4)))  _Float16 h4;
typedef __attribute__((ext_vector_type(8)))  float    v8f;

#define EMB   1024
#define SEQ   2048
#define BATCH 4
#define HEADS 16
#define HD    64
#define NROWS (BATCH * SEQ)     // 8192
#define ATTN_SCALE 0.125f       // 1/sqrt(64)

__device__ __forceinline__ v8f wmma16(v16h a, v16h b, v8f c) {
  // D = A(16x32 f16) * B(32x16 f16) + C(16x16 f32)
  return __builtin_amdgcn_wmma_f32_16x16x32_f16(
      /*neg_a=*/false, a, /*neg_b=*/false, b,
      /*c_mod=*/(short)0, c, /*reuse_a=*/false, /*reuse_b=*/false);
}

__device__ __forceinline__ void async_copy16(const _Float16* gsrc, _Float16* ldst) {
  const uint64_t ga = (uint64_t)(uintptr_t)gsrc;
  const uint32_t la = (uint32_t)(uintptr_t)ldst;
  asm volatile("global_load_async_to_lds_b128 %0, %1, off"
               :: "v"(la), "v"(ga) : "memory");
}

// ---------------------------------------------------------------------------
// Stage 0: fp32 -> fp16 conversion (vectorized x4)
// ---------------------------------------------------------------------------
__global__ __launch_bounds__(256) void cvt_f32_to_f16(
    const float* __restrict__ src, _Float16* __restrict__ dst, int n4) {
  int i = blockIdx.x * blockDim.x + threadIdx.x;
  if (i < n4) {
    float4 v = ((const float4*)src)[i];
    h4 o;
    o.x = (_Float16)v.x; o.y = (_Float16)v.y;
    o.z = (_Float16)v.z; o.w = (_Float16)v.w;
    ((h4*)dst)[i] = o;
  }
}

// ---------------------------------------------------------------------------
// Stage 1: QKV projection GEMM (NT): out[n,e] = sum_k X[n,k] * W[e,k] + b[e]
// One wave -> 64x64 output tile (16 WMMA accumulators, 4 A-frags x 4 B-frags
// per 32-wide k-step -> 64 B operand traffic per output element).
// Q,K stored [B,H,S,D] f16; V stored transposed [B,H,D,S] f16.
// ---------------------------------------------------------------------------
__global__ __launch_bounds__(256) void qkv_gemm(
    const _Float16* __restrict__ Xh, const _Float16* __restrict__ Wh,
    const float* __restrict__ pbq, const float* __restrict__ pbk,
    const float* __restrict__ pbv,
    _Float16* __restrict__ Qh, _Float16* __restrict__ Kh,
    _Float16* __restrict__ Vt) {
  const int lane = threadIdx.x & 31;
  const int half = lane >> 4;          // 0: lanes 0-15, 1: lanes 16-31
  const int lm   = lane & 15;
  const int wid  = blockIdx.x * 8 + (threadIdx.x >> 5);
  const int w    = wid >> 11;          // matrix: 0=Q 1=K 2=V (2048 jobs each)
  const int rem  = wid & 2047;
  const int n0   = (rem >> 4) << 6;    // 128 row tiles of 64
  const int e0   = (rem & 15) << 6;    // 16 col strips of 64

  const _Float16* wbase = Wh + (size_t)w * EMB * EMB;
  const float*    bias  = (w == 0) ? pbq : ((w == 1) ? pbk : pbv);

  const _Float16* xrow[4];
  const _Float16* wrow[4];
#pragma unroll
  for (int R = 0; R < 4; ++R) xrow[R] = Xh + (size_t)(n0 + R * 16 + lm) * EMB;
#pragma unroll
  for (int S = 0; S < 4; ++S) wrow[S] = wbase + (size_t)(e0 + S * 16 + lm) * EMB;

  v8f acc[4][4] = {};
  for (int kb = 0; kb < EMB; kb += 32) {
    if (kb + 64 < EMB) {
      __builtin_prefetch(xrow[0] + kb + 64, 0, 0);
      __builtin_prefetch(wrow[0] + kb + 64, 0, 0);
    }
    union { v16h v; h8 h[2]; } a[4];
    v16h bf[4];
#pragma unroll
    for (int R = 0; R < 4; ++R) {
      // A-frag (16x32 f16): lane = row, k contiguous at half*16
      a[R].h[0] = *(const h8*)(xrow[R] + kb + half * 16);
      a[R].h[1] = *(const h8*)(xrow[R] + kb + half * 16 + 8);
    }
#pragma unroll
    for (int S = 0; S < 4; ++S)
      // B-frag (32x16 f16): lane = col (W row), k contiguous at half*16
      bf[S] = *(const v16h*)(wrow[S] + kb + half * 16);
#pragma unroll
    for (int R = 0; R < 4; ++R)
#pragma unroll
      for (int S = 0; S < 4; ++S)
        acc[R][S] = wmma16(a[R].v, bf[S], acc[R][S]);
  }

#pragma unroll
  for (int S = 0; S < 4; ++S) {
    const int e  = e0 + S * 16 + lm;   // output column
    const float be = bias[e];
    const int hh = e >> 6, d = e & 63;
#pragma unroll
    for (int R = 0; R < 4; ++R)
#pragma unroll
      for (int r = 0; r < 8; ++r) {
        const int gr = n0 + R * 16 + r + half * 8;   // global row
        const int bb = gr >> 11, s = gr & 2047;      // batch, seq
        const _Float16 hv = (_Float16)(acc[R][S][r] + be);
        if (w == 2)
          Vt[(size_t)((bb * HEADS + hh) * HD + d) * SEQ + s] = hv;
        else
          ((w == 0) ? Qh : Kh)[(size_t)((bb * HEADS + hh) * SEQ + s) * HD + d] = hv;
      }
  }
}

// ---------------------------------------------------------------------------
// Stage 2: flash attention. 8 waves/block share one (b,h) and 128 consecutive
// queries. K/V 32-key tiles are double-buffered in LDS; both are staged with
// CDNA5 async global->LDS copies (ASYNCcnt) issued one block ahead, so the
// fetch of block i+1 is hidden under the 8 WMMAs + softmax of block i.
// ---------------------------------------------------------------------------
struct alignas(64) WaveScratch {
  float    sc[16 * 32];   // score tile (f32, row-major)
  float    fac[16];       // per-row broadcast (corr factor / row sum)
  _Float16 p[16 * 32];    // probabilities (f16, row-major) -> A-frag source
};

__global__ __launch_bounds__(256) void flash_attn(
    const _Float16* __restrict__ Qh, const _Float16* __restrict__ Kh,
    const _Float16* __restrict__ Vt, const int* __restrict__ pad,
    float* __restrict__ out) {
  __shared__ _Float16 Klds[2][32 * 64];   // [buf][key_local][dim]
  __shared__ _Float16 Vlds[2][64 * 32];   // [buf][dim][key_local]
  __shared__ WaveScratch smem[8];

  const int tid    = threadIdx.x;
  const int lane   = tid & 31;
  const int wavein = tid >> 5;
  WaveScratch* L   = &smem[wavein];
  const int bh   = blockIdx.x >> 4;          // B*H = 64
  const int base = (blockIdx.x & 15) << 7;   // 128-query super-tile
  const int q0   = base + wavein * 16;       // this wave's query tile
  const int b    = bh >> 4, h = bh & 15;
  const int half = lane >> 4, lm = lane & 15;

  // staging assignments (one 16B chunk per thread per tile)
  const int krow = tid >> 3, kch = tid & 7;   // K: 32 rows x 8 chunks
  const int vrow = tid >> 2, vch = tid & 3;   // V: 64 rows x 4 chunks
  const _Float16* kgsrc = Kh + ((size_t)bh * SEQ + krow) * HD + kch * 8;
  const _Float16* vgsrc = Vt + ((size_t)bh * HD + vrow) * SEQ + vch * 8;
  _Float16* kldst0 = &Klds[0][krow * 64 + kch * 8];
  _Float16* vldst0 = &Vlds[0][vrow * 32 + vch * 8];

  // Q A-frags for k-dims [0,32) and [32,64)
  const _Float16* qrow = Qh + (size_t)(bh * SEQ + q0 + lm) * HD;
  union { v16h v; h8 h2[2]; } aq0, aq1;
  aq0.h2[0] = *(const h8*)(qrow + half * 16);
  aq0.h2[1] = *(const h8*)(qrow + half * 16 + 8);
  aq1.h2[0] = *(const h8*)(qrow + 32 + half * 16);
  aq1.h2[1] = *(const h8*)(qrow + 40 + half * 16);

  v8f o0 = {}, o1 = {}, o2 = {}, o3 = {};
  float rm = -INFINITY, rl = 0.f;

  const int nb = (base + 128) >> 5;   // uniform causal trip count for block

  // prologue: stage block 0 into buffer 0
  async_copy16(kgsrc, kldst0);                       // K rows: kb offset 0
  async_copy16(vgsrc, vldst0);                       // V rows: kb offset 0

  for (int blk = 0; blk < nb; ++blk) {
    const int kb  = blk * 32;
    const int cur = blk & 1;

    // my stage-blk asyncs are the only outstanding ones -> drain them,
    // then barrier makes the whole tile visible (and proves all waves are
    // done computing on the other buffer, which we overwrite next).
    asm volatile("s_wait_asynccnt 0" ::: "memory");
    __syncthreads();

    if (blk + 1 < nb) {   // prefetch next block into the other buffer
      const int nkb = kb + 32;
      async_copy16(kgsrc + (size_t)nkb * HD, &Klds[cur ^ 1][krow * 64 + kch * 8]);
      async_copy16(vgsrc + nkb,              &Vlds[cur ^ 1][vrow * 32 + vch * 8]);
    }

    if (kb < q0 + 16) {   // causally live for this wave
      // ---- scores: two 16-key tiles of QK^T, K fragments from LDS ----
      const _Float16* Kbuf = Klds[cur];
      for (int t = 0; t < 2; ++t) {
        const int key = kb + t * 16 + lm;   // this lane's key column
        const _Float16* krowl = Kbuf + (t * 16 + lm) * 64;
        v16h bk0 = *(const v16h*)(krowl + half * 16);        // dims [0,32)
        v16h bk1 = *(const v16h*)(krowl + 32 + half * 16);   // dims [32,64)
        v8f s = {};
        s = wmma16(aq0.v, bk0, s);
        s = wmma16(aq1.v, bk1, s);
        const bool pk = pad[b * SEQ + key] != 0;
        for (int r = 0; r < 8; ++r) {
          const int qi = q0 + r + half * 8;
          float sv = s[r] * ATTN_SCALE;
          if (!pk || key > qi) sv = -INFINITY;
          L->sc[(r + half * 8) * 32 + t * 16 + lm] = sv;
        }
      }
      asm volatile("s_wait_dscnt 0" ::: "memory");

      // ---- online softmax, one row per lane (halves duplicate) ----
      {
        const float* srow = &L->sc[lm * 32];
        float mx = rm;
        for (int j = 0; j < 32; ++j) mx = fmaxf(mx, srow[j]);
        const float safem = fmaxf(mx, -1e30f);   // all-masked row: exp(-inf)=0
        const float corr  = __expf(rm - safem);
        float sum = 0.f;
        _Float16* prow = &L->p[lm * 32];
        for (int j = 0; j < 32; ++j) {
          const float pv = __expf(srow[j] - safem);
          sum += pv;
          prow[j] = (_Float16)pv;
        }
        rl = rl * corr + sum;
        rm = mx;
        L->fac[lm] = corr;
      }
      asm volatile("s_wait_dscnt 0" ::: "memory");

      // ---- rescale accumulators ----
      float fr[8];
      for (int r = 0; r < 8; ++r) fr[r] = L->fac[r + half * 8];
      for (int r = 0; r < 8; ++r) {
        o0[r] *= fr[r]; o1[r] *= fr[r]; o2[r] *= fr[r]; o3[r] *= fr[r];
      }

      // ---- P A-frag from LDS (C->A layout transpose happens here) ----
      union { v16h v; h8 h2[2]; } pf;
      const _Float16* prow = &L->p[lm * 32];
      pf.h2[0] = *(const h8*)(prow + half * 16);
      pf.h2[1] = *(const h8*)(prow + half * 16 + 8);

      // ---- V B-frags from LDS ([dim][key]: keys contiguous) ----
      const _Float16* Vbuf = Vlds[cur];
      v16h bv0 = *(const v16h*)(Vbuf + ( 0 + lm) * 32 + half * 16);
      v16h bv1 = *(const v16h*)(Vbuf + (16 + lm) * 32 + half * 16);
      v16h bv2 = *(const v16h*)(Vbuf + (32 + lm) * 32 + half * 16);
      v16h bv3 = *(const v16h*)(Vbuf + (48 + lm) * 32 + half * 16);
      o0 = wmma16(pf.v, bv0, o0);
      o1 = wmma16(pf.v, bv1, o1);
      o2 = wmma16(pf.v, bv2, o2);
      o3 = wmma16(pf.v, bv3, o3);
    }
  }

  // ---- finalize: broadcast row sums, divide, zero padded rows, store ----
  asm volatile("s_wait_dscnt 0" ::: "memory");
  L->fac[lm] = rl;
  asm volatile("s_wait_dscnt 0" ::: "memory");
  float inv[8];
  for (int r = 0; r < 8; ++r) {
    const float l = L->fac[r + half * 8];
    inv[r] = (l > 0.f) ? 1.f / l : 0.f;
  }
  for (int r = 0; r < 8; ++r) {
    const int gq = q0 + r + half * 8;
    if (pad[b * SEQ + gq] == 0) inv[r] = 0.f;   // fully-masked row -> 0
  }
  const size_t obase = (size_t)b * SEQ * EMB + (size_t)h * HD + lm;
  for (int r = 0; r < 8; ++r) {
    const int gq = q0 + r + half * 8;
    float* orow = out + obase + (size_t)gq * EMB;
    orow[0]  = o0[r] * inv[r];
    orow[16] = o1[r] * inv[r];
    orow[32] = o2[r] * inv[r];
    orow[48] = o3[r] * inv[r];
  }
}

// ---------------------------------------------------------------------------
extern "C" void kernel_launch(void* const* d_in, const int* in_sizes, int n_in,
                              void* d_out, int out_size, void* d_ws, size_t ws_size,
                              hipStream_t stream) {
  const float* X   = (const float*)d_in[0];
  const int*   pad = (const int*)d_in[1];
  const float* Wq  = (const float*)d_in[2];
  const float* bq  = (const float*)d_in[3];
  const float* Wk  = (const float*)d_in[4];
  const float* bk  = (const float*)d_in[5];
  const float* Wv  = (const float*)d_in[6];
  const float* bv  = (const float*)d_in[7];
  float* out = (float*)d_out;

  const size_t NX = (size_t)NROWS * EMB;   // 8192*1024
  const size_t NW = (size_t)EMB * EMB;     // 1024*1024
  _Float16* Xh = (_Float16*)d_ws;
  _Float16* Wh = Xh + NX;
  _Float16* Qh = Wh + 3 * NW;
  _Float16* Kh = Qh + NX;
  _Float16* Vt = Kh + NX;

  int n4 = (int)(NX / 4);
  cvt_f32_to_f16<<<(n4 + 255) / 256, 256, 0, stream>>>(X, Xh, n4);
  n4 = (int)(NW / 4);
  cvt_f32_to_f16<<<(n4 + 255) / 256, 256, 0, stream>>>(Wq, Wh + 0 * NW, n4);
  cvt_f32_to_f16<<<(n4 + 255) / 256, 256, 0, stream>>>(Wk, Wh + 1 * NW, n4);
  cvt_f32_to_f16<<<(n4 + 255) / 256, 256, 0, stream>>>(Wv, Wh + 2 * NW, n4);

  // 3 matrices * 128 row-tiles * 16 col-strips = 6144 waves / 8 per block
  qkv_gemm<<<768, 256, 0, stream>>>(Xh, Wh, bq, bk, bv, Qh, Kh, Vt);

  // 64 (B*H) * 16 query super-tiles = 1024 blocks of 8 waves
  flash_attn<<<1024, 256, 0, stream>>>(Qh, Kh, Vt, pad, out);
}
